// HybridAttention_74483322847371
// MI455X (gfx1250) — matrix operand order repaired and checked by reading, not compiled, not run on hardware
//
#include <hip/hip_runtime.h>

// ---------------------------------------------------------------------------
// Swin window attention for MI455X (gfx1250), bf16 WMMA pipeline.
// B_=2048 windows, N=64 tokens, C=256, H=8 heads, HD=32.
// Weights pre-packed into WMMA-B fragment layout so every fragment load is
// two b128 accesses (no scalar u16 gathers).
// ---------------------------------------------------------------------------

typedef __attribute__((ext_vector_type(16))) __bf16 v16bf;
typedef __attribute__((ext_vector_type(8)))  __bf16 v8bf;
typedef __attribute__((ext_vector_type(8)))  float  v8f;

__device__ __forceinline__ __bf16 f2bf(float f) {
    unsigned u = __builtin_bit_cast(unsigned, f);
    unsigned r = (u + 0x7FFFu + ((u >> 16) & 1u)) >> 16;   // round-nearest-even
    unsigned short s = (unsigned short)r;
    return __builtin_bit_cast(__bf16, s);
}
__device__ __forceinline__ unsigned pack2bf(float a, float b) {
    unsigned lo = __builtin_bit_cast(unsigned short, f2bf(a));
    unsigned hi = __builtin_bit_cast(unsigned short, f2bf(b));
    return lo | (hi << 16);
}
__device__ __forceinline__ v16bf cat16(v8bf lo, v8bf hi) {
    return __builtin_shufflevector(lo, hi, 0,1,2,3,4,5,6,7,8,9,10,11,12,13,14,15);
}

#define WMMA_BF16(A, B, C) \
    __builtin_amdgcn_wmma_f32_16x16x32_bf16(false, (A), false, (B), (short)0, (C), false, false)

// ---------------------------------------------------------------------------
// Kernel 0: preparation.
//  - wqkvF: [12 nblocks][8 ksteps][4 ntiles][32 lanes][16] bf16 fragment pack
//           (q-scale folded into wq columns)
//  - wpF:   [ 4 nblocks][8 ksteps][4 ntiles][32 lanes][16] bf16 fragment pack
//  - bqkv:  [768] fp32 (bq scaled)
//  - biasHNM: [8][64][64] fp32 relative-position bias
// Fragment element map: k = ks*32 + (lane>>4)*16 + e ; n = nb*64+nt*16+(lane&15)
// ---------------------------------------------------------------------------
__global__ void prep_kernel(const float* __restrict__ wq, const float* __restrict__ wk,
                            const float* __restrict__ wv, const float* __restrict__ wp,
                            const float* __restrict__ bq, const float* __restrict__ bk,
                            const float* __restrict__ bv,
                            const float* __restrict__ bias_table, const int* __restrict__ rel_index,
                            __bf16* __restrict__ wqkvF, __bf16* __restrict__ wpF,
                            float* __restrict__ bqkv, float* __restrict__ biasHNM) {
    const float SCALE = 0.17677669529663687f;   // 32^-0.5 folded into q path
    int t = blockIdx.x * 256 + threadIdx.x;
    if (t < 196608) {                           // wqkvF
        int e = t & 15, lane = (t >> 4) & 31, nt = (t >> 9) & 3, ks = (t >> 11) & 7, nb = t >> 14;
        int k = ks * 32 + (lane >> 4) * 16 + e;
        int col = nb * 64 + nt * 16 + (lane & 15);
        int sel = col >> 8, c = col & 255;
        float v = (sel == 0) ? wq[k * 256 + c] * SCALE
                : (sel == 1) ? wk[k * 256 + c]
                             : wv[k * 256 + c];
        wqkvF[t] = f2bf(v);
    } else if (t < 262144) {                    // wpF
        int j = t - 196608;
        int e = j & 15, lane = (j >> 4) & 31, nt = (j >> 9) & 3, ks = (j >> 11) & 7, nb = j >> 14;
        int k = ks * 32 + (lane >> 4) * 16 + e;
        int n = nb * 64 + nt * 16 + (lane & 15);
        wpF[j] = f2bf(wp[k * 256 + n]);
    } else if (t < 262912) {                    // bqkv
        int j = t - 262144;
        bqkv[j] = (j < 256) ? bq[j] * SCALE : (j < 512) ? bk[j - 256] : bv[j - 512];
    } else if (t < 295680) {                    // biasHNM
        int j = t - 262912;
        int h = j >> 12, nm = j & 4095;
        biasHNM[j] = bias_table[rel_index[nm] * 8 + h];
    }
}

// ---------------------------------------------------------------------------
// Kernel 1: QKV projection.  (2048*64)x768 = x(131072x256) @ wqkv(256x768).
// Block: 128 threads (4 waves).  Stage full 64x256 A panel once; weights
// staged 16KB at a time from the fragment-packed buffer.
// ---------------------------------------------------------------------------
__global__ void qkv_gemm(const float* __restrict__ x, const __bf16* __restrict__ wqkvF,
                         const float* __restrict__ bqkv, __bf16* __restrict__ qkv) {
    __shared__ alignas(16) __bf16 Xs[64][264];   // 528B rows: 16B-aligned, 132-bank stride
    __shared__ alignas(16) __bf16 WF[8192];      // 16 fragments (4 ksteps x 4 ntiles)
    const int b  = blockIdx.x;                   // window
    const int nb = blockIdx.y;                   // 0..11 -> output cols 64*nb
    const int t = threadIdx.x;
    const int lane = t & 31, wave = t >> 5;
    const int mBase = wave * 16;
    const int hi = lane >> 4, lo = lane & 15;

    // stage A panel: 64x256 fp32 -> bf16
    #pragma unroll 8
    for (int j = 0; j < 32; ++j) {
        int idx = t + 128 * j;                   // 0..4095 float4s
        int r = idx >> 6, c4 = (idx & 63) * 4;
        const float4 v = *(const float4*)(x + ((size_t)(b * 64 + r)) * 256 + c4);
        *(unsigned*)&Xs[r][c4 + 0] = pack2bf(v.x, v.y);
        *(unsigned*)&Xs[r][c4 + 2] = pack2bf(v.z, v.w);
    }

    v8f acc[4] = {};
    for (int half = 0; half < 2; ++half) {
        // stage 16KB of fragment-packed weights (ksteps half*4 .. half*4+3)
        const uint4* wsrc = (const uint4*)(wqkvF + (size_t)nb * 16384 + half * 8192);
        uint4* wdst = (uint4*)WF;
        #pragma unroll
        for (int j = 0; j < 8; ++j) wdst[t + 128 * j] = wsrc[t + 128 * j];
        __syncthreads();

        #pragma unroll
        for (int ks = 0; ks < 4; ++ks) {
            const int kb = (half * 4 + ks) * 32 + 16 * hi;
            const int r = mBase + lo;
            v16bf a = cat16(*(const v8bf*)&Xs[r][kb], *(const v8bf*)&Xs[r][kb + 8]);
            #pragma unroll
            for (int nt = 0; nt < 4; ++nt) {
                const __bf16* p = WF + ((ks * 4 + nt) * 32 + lane) * 16;
                v16bf bb = cat16(*(const v8bf*)p, *(const v8bf*)(p + 8));
                acc[nt] = WMMA_BF16(a, bb, acc[nt]);
            }
        }
        __syncthreads();
    }

    #pragma unroll
    for (int nt = 0; nt < 4; ++nt) {
        const int col = nb * 64 + nt * 16 + lo;
        const float bias = bqkv[col];
        #pragma unroll
        for (int i = 0; i < 8; ++i) {
            const int row = mBase + i + 8 * hi;
            qkv[((size_t)(b * 64 + row)) * 768 + col] = f2bf(acc[nt][i] + bias);
        }
    }
}

// ---------------------------------------------------------------------------
// Kernel 2: per (window, head) attention.
// scores = Q Kt (+bias+mask) -> softmax -> P V.  Output overwrites the q
// columns of qkv (each block only touches its own head's 32 columns).
// ---------------------------------------------------------------------------
__global__ void attn_kernel(__bf16* qkv, const float* __restrict__ biasHNM,
                            const float* __restrict__ mask) {
    __shared__ alignas(16) __bf16 Qs[64][40];    // 80B rows (16B aligned)
    __shared__ alignas(16) __bf16 Ks[64][40];
    __shared__ alignas(16) __bf16 VT[32][72];    // V transposed: VT[d][m], 144B rows
    __shared__ alignas(16) float  Sf[64][65];
    __shared__ alignas(16) __bf16 Pb[64][72];
    __shared__ float redA[2][64], redB[2][64];
    const int b = blockIdx.x, h = blockIdx.y;
    const int nw = b & 63;                       // window-in-image index
    const int t = threadIdx.x;
    const int lane = t & 31, wave = t >> 5;
    const int mBase = wave * 16;
    const int hi = lane >> 4, lo = lane & 15;

    // stage Q/K (row major) and V (transposed)
    #pragma unroll
    for (int j = 0; j < 8; ++j) {
        int idx = t + 128 * j;                   // 0..1023
        int n = idx >> 4, du = idx & 15;
        const unsigned* src = (const unsigned*)(qkv + ((size_t)(b * 64 + n)) * 768 + h * 32);
        ((unsigned*)&Qs[n][0])[du] = src[du];
        ((unsigned*)&Ks[n][0])[du] = src[du + 128];   // k at +256 bf16
        unsigned vv = src[du + 256];                   // v at +512 bf16
        VT[du * 2 + 0][n] = __builtin_bit_cast(__bf16, (unsigned short)(vv & 0xFFFFu));
        VT[du * 2 + 1][n] = __builtin_bit_cast(__bf16, (unsigned short)(vv >> 16));
    }
    __syncthreads();

    // S = Q @ K^T   (wave: 16 rows x 64 cols; single K-step of 32)
    v8f s[4] = {};
    {
        const int r = mBase + lo;
        const int kb = 16 * hi;
        v16bf a = cat16(*(const v8bf*)&Qs[r][kb], *(const v8bf*)&Qs[r][kb + 8]);
        #pragma unroll
        for (int nt = 0; nt < 4; ++nt) {
            const int rn = nt * 16 + lo;          // B[k][n] = K[n][k]: contiguous k
            v16bf bb = cat16(*(const v8bf*)&Ks[rn][16 * hi],
                             *(const v8bf*)&Ks[rn][16 * hi + 8]);
            s[nt] = WMMA_BF16(a, bb, s[nt]);
        }
    }
    // + relpos bias + window mask -> Sf
    #pragma unroll
    for (int nt = 0; nt < 4; ++nt) {
        #pragma unroll
        for (int i = 0; i < 8; ++i) {
            const int row = mBase + i + 8 * hi;
            const int col = nt * 16 + lo;
            Sf[row][col] = s[nt][i] + biasHNM[h * 4096 + row * 64 + col]
                                    + mask[(size_t)nw * 4096 + row * 64 + col];
        }
    }
    __syncthreads();

    // fp32 softmax: 2 threads per row (halves), LDS reductions
    {
        const int r = t & 63, hf = t >> 6;
        float* row = &Sf[r][hf * 32];
        float mx = -3.0e38f;
        #pragma unroll 8
        for (int c = 0; c < 32; ++c) mx = fmaxf(mx, row[c]);
        redA[hf][r] = mx;
        __syncthreads();
        mx = fmaxf(redA[0][r], redA[1][r]);
        float sum = 0.f;
        #pragma unroll 8
        for (int c = 0; c < 32; ++c) { float e = __expf(row[c] - mx); row[c] = e; sum += e; }
        redB[hf][r] = sum;
        __syncthreads();
        const float inv = 1.0f / (redB[0][r] + redB[1][r]);
        #pragma unroll 8
        for (int c = 0; c < 32; ++c) Pb[r][hf * 32 + c] = f2bf(row[c] * inv);
    }
    __syncthreads();

    // O = P @ V   (wave: 16 rows x 32 cols, two K-steps)
    v8f o[2] = {};
    #pragma unroll
    for (int ks = 0; ks < 2; ++ks) {
        const int r = mBase + lo;
        const int kb = ks * 32 + 16 * hi;
        v16bf a = cat16(*(const v8bf*)&Pb[r][kb], *(const v8bf*)&Pb[r][kb + 8]);
        #pragma unroll
        for (int nt = 0; nt < 2; ++nt) {
            const int d = nt * 16 + lo;           // B[k=m][n=d] = VT[d][m]: contiguous m
            const int m0 = ks * 32 + 16 * hi;
            v16bf bb = cat16(*(const v8bf*)&VT[d][m0], *(const v8bf*)&VT[d][m0 + 8]);
            o[nt] = WMMA_BF16(a, bb, o[nt]);
        }
    }
    // store into the q columns of qkv (head-disjoint across blocks)
    #pragma unroll
    for (int nt = 0; nt < 2; ++nt) {
        #pragma unroll
        for (int i = 0; i < 8; ++i) {
            const int row = mBase + i + 8 * hi;
            const int col = nt * 16 + lo;
            qkv[((size_t)(b * 64 + row)) * 768 + h * 32 + col] = f2bf(o[nt][i]);
        }
    }
}

// ---------------------------------------------------------------------------
// Kernel 3: output projection. (2048*64)x256 = attn(131072x256) @ wp.
// ---------------------------------------------------------------------------
__global__ void proj_gemm(const __bf16* __restrict__ qkv, const __bf16* __restrict__ wpF,
                          const float* __restrict__ bp, float* __restrict__ out) {
    __shared__ alignas(16) __bf16 Xs[64][264];
    __shared__ alignas(16) __bf16 WF[8192];
    const int b  = blockIdx.x;
    const int nb = blockIdx.y;                   // 0..3
    const int t = threadIdx.x;
    const int lane = t & 31, wave = t >> 5;
    const int mBase = wave * 16;
    const int hi = lane >> 4, lo = lane & 15;

    // stage A panel (attention output, bf16, q columns of qkv)
    #pragma unroll 8
    for (int j = 0; j < 16; ++j) {
        int idx = t + 128 * j;                   // 0..2047 uint4s (8 bf16 each)
        int r = idx >> 5, c8 = (idx & 31) * 8;
        const uint4 v = *(const uint4*)(qkv + ((size_t)(b * 64 + r)) * 768 + c8);
        *(uint4*)&Xs[r][c8] = v;
    }

    v8f acc[4] = {};
    for (int half = 0; half < 2; ++half) {
        const uint4* wsrc = (const uint4*)(wpF + (size_t)nb * 16384 + half * 8192);
        uint4* wdst = (uint4*)WF;
        #pragma unroll
        for (int j = 0; j < 8; ++j) wdst[t + 128 * j] = wsrc[t + 128 * j];
        __syncthreads();

        #pragma unroll
        for (int ks = 0; ks < 4; ++ks) {
            const int kb = (half * 4 + ks) * 32 + 16 * hi;
            const int r = mBase + lo;
            v16bf a = cat16(*(const v8bf*)&Xs[r][kb], *(const v8bf*)&Xs[r][kb + 8]);
            #pragma unroll
            for (int nt = 0; nt < 4; ++nt) {
                const __bf16* p = WF + ((ks * 4 + nt) * 32 + lane) * 16;
                v16bf bb = cat16(*(const v8bf*)p, *(const v8bf*)(p + 8));
                acc[nt] = WMMA_BF16(a, bb, acc[nt]);
            }
        }
        __syncthreads();
    }

    #pragma unroll
    for (int nt = 0; nt < 4; ++nt) {
        const int col = nb * 64 + nt * 16 + lo;
        const float bias = bp[col];
        #pragma unroll
        for (int i = 0; i < 8; ++i) {
            const int row = mBase + i + 8 * hi;
            out[((size_t)(b * 64 + row)) * 256 + col] = acc[nt][i] + bias;
        }
    }
}

// ---------------------------------------------------------------------------
// Launcher.  ws layout (bytes):
//   [0,        393216)  wqkvF bf16 fragment-packed [12][8][4][32][16]
//   [393216,   524288)  wpF   bf16 fragment-packed [ 4][8][4][32][16]
//   [524288,   527360)  bqkv f32  [768]
//   [527360,   658432)  biasHNM f32 [8][64][64]
//   [658432, 201985024) qkv  bf16 [2048*64][768]  (q cols reused as attn out)
// ---------------------------------------------------------------------------
extern "C" void kernel_launch(void* const* d_in, const int* in_sizes, int n_in,
                              void* d_out, int out_size, void* d_ws, size_t ws_size,
                              hipStream_t stream) {
    const float* x          = (const float*)d_in[0];
    const float* mask       = (const float*)d_in[1];
    const float* wq         = (const float*)d_in[2];
    const float* bq         = (const float*)d_in[3];
    const float* wk         = (const float*)d_in[4];
    const float* bk         = (const float*)d_in[5];
    const float* wv         = (const float*)d_in[6];
    const float* bv         = (const float*)d_in[7];
    const float* wp         = (const float*)d_in[8];
    const float* bp         = (const float*)d_in[9];
    const float* bias_table = (const float*)d_in[10];
    const int*   rel_index  = (const int*)d_in[11];

    char* ws = (char*)d_ws;
    __bf16* wqkvF   = (__bf16*)(ws + 0);
    __bf16* wpF     = (__bf16*)(ws + 393216);
    float*  bqkv    = (float*)(ws + 524288);
    float*  biasHNM = (float*)(ws + 527360);
    __bf16* qkv     = (__bf16*)(ws + 658432);

    prep_kernel<<<1155, 256, 0, stream>>>(wq, wk, wv, wp, bq, bk, bv,
                                          bias_table, rel_index,
                                          wqkvF, wpF, bqkv, biasHNM);
    qkv_gemm<<<dim3(2048, 12), 128, 0, stream>>>(x, wqkvF, bqkv, qkv);
    attn_kernel<<<dim3(2048, 8), 128, 0, stream>>>(qkv, biasHNM, mask);
    proj_gemm<<<dim3(2048, 4), 128, 0, stream>>>(qkv, wpF, bp, (float*)d_out);
}
